// QuadrupletInteraction_69999376990650
// MI455X (gfx1250) — matrix operand order repaired and checked by reading, not compiled, not run on hardware
//
#include <hip/hip_runtime.h>
#include <math.h>

// ---------------------------------------------------------------------------
// GemNet quadruplet interaction for MI455X (gfx1250, wave32, WMMA).
//
//   k_swz       : f32 -> bf16 weights, fragment-major (one contiguous 32B
//                 vector load per lane per B-fragment), optional K zero-pad
//   k_edge      : x_down = silu( (silu(m@W_db) * (rad@W_rbf)*s_rbf) @ W_down )
//                 fully fused; both GEMMs *and* the rbf gate run on WMMA
//   k_quad      : per 16-edge tile: gather quad slots, recompute triplet rows
//                 on the fly, 7-wide einsums in VALU, bilinear via WMMA
//   k_up        : out = (silu(x@W_up_ca) + silu(x@W_up_ac)[id_swap]) / sqrt(2)
// ---------------------------------------------------------------------------

typedef __attribute__((ext_vector_type(16))) __bf16 v16bf;
typedef __attribute__((ext_vector_type(4)))  __bf16 v4bf;
typedef __attribute__((ext_vector_type(8)))  float  v8f;

#define E_DIM   512
#define QIN     32
#define QOUT    32
#define RBF     16
#define CBF     16
#define SBF     32
#define NSPH    7
#define KAVG    20

// silu via v_rcp_f32 instead of the IEEE div expansion (~4 VALU vs ~12).
__device__ inline float silu_f(float x) {
  return x * __builtin_amdgcn_rcpf(1.0f + __expf(-x));
}

// --- WMMA fragment helpers (layouts per CDNA5 ISA §7.12.2) -------

// A-matrix 16x32 bf16 from row-major LDS, leading dim `ld` (ld*2 must be a
// multiple of 16B so the 8-element runs lower to aligned ds_load_b128).
// lanes 0-15: K 0..7 (v0-3), 16..23 (v4-7); lanes 16-31: K 8..15 / 24..31.
__device__ inline v16bf load_A_frag(const __bf16* base, int ld, int lane) {
  const int half = lane >> 4;
  const int row  = lane & 15;
  v16bf a;
#pragma unroll
  for (int t = 0; t < 16; ++t) {
    const int v  = t >> 1;
    const int hi = t & 1;
    const int k  = ((v >> 2) << 4) + (half << 3) + ((v & 3) << 1) + hi;
    a[t] = base[row * ld + k];
  }
  return a;
}

// B-matrix fragment from swizzled weights: tile (kt,nt) of a KxN matrix is
// stored as 32 lanes x 16 contiguous bf16 (lane-major, 32B aligned).
__device__ inline v16bf load_B_swz(const __bf16* __restrict__ swz,
                                   int kt, int nt, int ntiles, int lane) {
  const __bf16* p = swz + (((size_t)(kt * ntiles + nt)) << 9) + (lane << 4);
  return *(const v16bf*)p;
}

__device__ inline v8f wmma_bf16(v16bf a, v16bf b, v8f c) {
  return __builtin_amdgcn_wmma_f32_16x16x32_bf16(
      /*neg_a=*/false, a, /*neg_b=*/false, b,
      /*c_mod=*/(short)0, c, /*reuse_a=*/false, /*reuse_b=*/false);
}

// C/D element (m,n) for VGPR r: m = r + 8*(lane>>4), n = lane&15.

// ---------------------------------------------------------------------------
// Weight convert + swizzle with optional K zero-padding (rows >= Ksrc -> 0).
// out[i], i = ((kt*ntiles+nt)*32 + lane)*16 + t, holds
// W[kt*32 + (t>>3)*16 + (lane>>4)*8 + (t&7)][nt*16 + (lane&15)] as bf16 (kt over Kpad/32).
// ---------------------------------------------------------------------------
__global__ void k_swz(const float* __restrict__ W, __bf16* __restrict__ dst,
                      int Ksrc, int N, int Kpad) {
  const int i = blockIdx.x * blockDim.x + threadIdx.x;
  const int total = Kpad * N;
  if (i >= total) return;
  const int t    = i & 15;
  const int lane = (i >> 4) & 31;
  const int tile = i >> 9;
  const int ntiles = N >> 4;
  const int nt = tile % ntiles;
  const int kt = tile / ntiles;
  const int k = (kt << 5) + ((t >> 3) << 4) + ((lane >> 4) << 3) + (t & 7);
  const int n = (nt << 4) + (lane & 15);
  dst[i] = (k < Ksrc) ? (__bf16)W[(size_t)k * N + n] : (__bf16)0.0f;
}

// ---------------------------------------------------------------------------
// Fused edge MLP: silu(m@W_db) * (rad@W_rbf)*s_rbf -> @W_down -> silu
// 256 threads (8 waves), 64-row tile, full 512-wide activation tile in LDS.
// rbf gate is computed by WMMA too (K padded 16->32); its C layout matches
// the main accumulator layout exactly, so it fuses elementwise for free.
// ---------------------------------------------------------------------------
#define MT 64
#define XLD (E_DIM + 16)   // bf16 row stride; *2 = 1056B, multiple of 16B
#define RLD 48             // rad tile row stride (32 data + pad); 96B
__global__ __launch_bounds__(256) void k_edge(
    const float*  __restrict__ m,
    const float*  __restrict__ rad,
    const __bf16* __restrict__ Wdb_swz,    // 512x512, fragment-major
    const __bf16* __restrict__ Wrbf_swz,   // 32(pad)x512, fragment-major
    const __bf16* __restrict__ Wdown_swz,  // 512x32, fragment-major
    const float*  __restrict__ scale_rbf,
    float*        __restrict__ x_down,     // nE x 32
    int nE)
{
  __shared__ __bf16 s_x[MT][XLD];          // m tile, then reused as x1 tile
  __shared__ __bf16 s_radb[MT][RLD];       // rad tile, K-padded to 32, bf16

  const int tid  = threadIdx.x;
  const int wave = tid >> 5;
  const int lane = tid & 31;
  const int e0   = blockIdx.x * MT;

  // Load m tile (float4 global loads -> packed v4bf LDS stores).
  for (int i = tid; i < MT * (E_DIM / 4); i += 256) {
    const int r  = i >> 7;
    const int c4 = (i & 127) << 2;
    const int ge = e0 + r;
    float4 v = make_float4(0.f, 0.f, 0.f, 0.f);
    if (ge < nE) v = ((const float4*)m)[((size_t)ge * E_DIM + c4) >> 2];
    v4bf pk = { (__bf16)v.x, (__bf16)v.y, (__bf16)v.z, (__bf16)v.w };
    *(v4bf*)&s_x[r][c4] = pk;
  }
  // rad tile as bf16, columns 16..31 zero (K pad for the gate WMMA).
  for (int i = tid; i < MT * 32; i += 256) {
    const int r = i >> 5, c = i & 31;
    const int ge = e0 + r;
    float v = (ge < nE && c < RBF) ? rad[(size_t)ge * RBF + c] : 0.0f;
    s_radb[r][c] = (__bf16)v;
  }
  __syncthreads();

  // GEMM1: each wave owns 4 M-tiles x 4 N-tiles (cols wave*64 .. +63).
  v8f zero = {0.f, 0.f, 0.f, 0.f, 0.f, 0.f, 0.f, 0.f};
  v8f acc[4][4];
#pragma unroll
  for (int i = 0; i < 4; ++i)
#pragma unroll
    for (int j = 0; j < 4; ++j) acc[i][j] = zero;

  for (int kt = 0; kt < E_DIM / 32; ++kt) {
    const int k0 = kt << 5;
    v16bf a[4], b[4];
#pragma unroll
    for (int mi = 0; mi < 4; ++mi)
      a[mi] = load_A_frag(&s_x[mi * 16][k0], XLD, lane);
#pragma unroll
    for (int ni = 0; ni < 4; ++ni)
      b[ni] = load_B_swz(Wdb_swz, kt, wave * 4 + ni, E_DIM / 16, lane);
#pragma unroll
    for (int mi = 0; mi < 4; ++mi)
#pragma unroll
      for (int ni = 0; ni < 4; ++ni)
        acc[mi][ni] = wmma_bf16(a[mi], b[ni], acc[mi][ni]);
  }
  __syncthreads();

  // Epilogue 1: gate = rad@W_rbf via WMMA (K=32, one instr/tile), then
  // x1 = silu(acc) * gate * s_rbf, written back into s_x as bf16.
  const float srbf = scale_rbf[0];
  const int n_base = wave * 64;
  v16bf arad[4];
#pragma unroll
  for (int mi = 0; mi < 4; ++mi)
    arad[mi] = load_A_frag(&s_radb[mi * 16][0], RLD, lane);
#pragma unroll
  for (int ni = 0; ni < 4; ++ni) {
    v16bf brbf = load_B_swz(Wrbf_swz, 0, wave * 4 + ni, E_DIM / 16, lane);
#pragma unroll
    for (int mi = 0; mi < 4; ++mi) {
      v8f g = wmma_bf16(arad[mi], brbf, zero);
#pragma unroll
      for (int r = 0; r < 8; ++r) {
        const int mrow = mi * 16 + r + ((lane >> 4) << 3);
        const int ncol = n_base + ni * 16 + (lane & 15);
        const float v = silu_f(acc[mi][ni][r]) * g[r] * srbf;
        s_x[mrow][ncol] = (__bf16)v;
      }
    }
  }
  __syncthreads();

  // GEMM2: (64x512)@(512x32) -> 64x32; wave w owns tile (mi=w>>1, ni=w&1).
  const int mi2 = wave >> 1;
  const int ni2 = wave & 1;
  v8f c2 = zero;
  for (int kt = 0; kt < E_DIM / 32; ++kt) {
    v16bf a = load_A_frag(&s_x[mi2 * 16][kt << 5], XLD, lane);
    v16bf b = load_B_swz(Wdown_swz, kt, ni2, QIN / 16, lane);
    c2 = wmma_bf16(a, b, c2);
  }
#pragma unroll
  for (int r = 0; r < 8; ++r) {
    const int mrow = mi2 * 16 + r + ((lane >> 4) << 3);
    const int ncol = ni2 * 16 + (lane & 15);
    const int ge = e0 + mrow;
    if (ge < nE) x_down[(size_t)ge * QIN + ncol] = silu_f(c2[r]);
  }
}

// ---------------------------------------------------------------------------
// Quad kernel: 16 edges / block. Recompute triplet rows on gather, contract
// bases (7-wide, VALU), then bilinear (16x1024)@(1024x32) via WMMA.
// Ragged layout: edge e owns quads e*20 .. e*20+19 into slots 0..19 (of 32).
// ---------------------------------------------------------------------------
#define ET 16
#define RWLD (SBF * QIN + 8)   // *2 = 2064B, multiple of 16B
__global__ __launch_bounds__(256) void k_quad(
    const float*  __restrict__ x_down,          // nE x 32
    const float*  __restrict__ cir,             // nTrip x 16
    const float*  __restrict__ Wcbf,            // 16 x 32
    const float*  __restrict__ sph,             // nE x 32 x 7
    const float*  __restrict__ srw,             // nE x 32 x 7 (sph_rad_W1)
    const __bf16* __restrict__ Wbil_swz,        // 1024 x 32, fragment-major
    const float*  __restrict__ scale_cbf,
    const float*  __restrict__ scale_sbf,
    const int*    __restrict__ idx_trip_in,
    const int*    __restrict__ idx_trip_to_quad,
    float*        __restrict__ x_bil,           // nE x 32
    int nE)
{
  __shared__ float  s_g[ET][KAVG][QIN];         // 40 KB gathered triplet rows
  __shared__ float  s_sumk[ET][NSPH][QIN];      // 14 KB
  __shared__ __bf16 s_rw[ET][RWLD];             // 33 KB bf16 rW rows
  __shared__ float  s_wcbf[CBF][QIN];           //  2 KB
  __shared__ float  s_acc[ET][QOUT];            //  2 KB cross-wave reduce

  const int tid  = threadIdx.x;
  const int wave = tid >> 5;
  const int lane = tid & 31;
  const int e0   = blockIdx.x * ET;
  const float scbf = scale_cbf[0];
  const float ssbf = scale_sbf[0];

  for (int i = tid; i < CBF * QIN; i += 256) s_wcbf[i >> 5][i & 31] = Wcbf[i];
  for (int i = tid; i < ET * QOUT; i += 256) s_acc[i >> 5][i & 31] = 0.0f;
  __syncthreads();

  // Phase 1: gather quad slot -> triplet row, apply cbf gate.
  for (int idx = tid; idx < ET * KAVG * QIN; idx += 256) {
    const int d   = idx & 31;
    const int row = idx >> 5;            // 0..319, uniform across a wave
    const int e   = row / KAVG;
    const int k   = row - e * KAVG;
    const int ge  = e0 + e;
    float val = 0.0f;
    if (ge < nE) {
      const size_t q = (size_t)ge * KAVG + k;
      const int t   = idx_trip_to_quad[q];
      const int ein = idx_trip_in[t];
      const float xd = x_down[(size_t)ein * QIN + d];
      float cb = 0.0f;
#pragma unroll
      for (int j = 0; j < CBF; ++j) cb += cir[(size_t)t * CBF + j] * s_wcbf[j][d];
      val = xd * cb * scbf;
    }
    s_g[e][k][d] = val;
  }
  __syncthreads();

  // Phase 2: sum_k[e][s][d] = sum_{k<20} sph[e][k][s] * g[e][k][d]
  for (int idx = tid; idx < ET * NSPH * QIN; idx += 256) {
    const int d = idx & 31;
    const int s = (idx >> 5) % NSPH;
    const int e = (idx >> 5) / NSPH;
    const int ge = e0 + e;
    const float* sp = sph + ((size_t)min(ge, nE - 1) * 32) * NSPH + s;
    float acc = 0.0f;
#pragma unroll
    for (int k = 0; k < KAVG; ++k) acc += sp[k * NSPH] * s_g[e][k][d];
    s_sumk[e][s][d] = acc;
  }
  __syncthreads();

  // Phase 3: rW[e][i][d] = sum_s srw[e][i][s] * sum_k[e][s][d] -> bf16 LDS
  for (int idx = tid; idx < ET * SBF * QIN; idx += 256) {
    const int d = idx & 31;
    const int i = (idx >> 5) & 31;
    const int e = idx >> 10;
    const int ge = e0 + e;
    const float* sr = srw + ((size_t)min(ge, nE - 1) * SBF + i) * NSPH;
    float acc = 0.0f;
#pragma unroll
    for (int s = 0; s < NSPH; ++s) acc += sr[s] * s_sumk[e][s][d];
    s_rw[e][i * QIN + d] = (__bf16)acc;
  }
  __syncthreads();

  // Phase 4: X = RW(16x1024) @ W_bil(1024x32). 8 waves: 2 N-tiles x 4 K-chunks.
  const int ni = wave & 1;      // N-tile
  const int kc = wave >> 1;     // K-chunk of 256 (8 WMMA steps)
  v8f zero = {0.f, 0.f, 0.f, 0.f, 0.f, 0.f, 0.f, 0.f};
  v8f c = zero;
  for (int kt = kc * 8; kt < kc * 8 + 8; ++kt) {
    v16bf a = load_A_frag(&s_rw[0][kt << 5], RWLD, lane);
    v16bf b = load_B_swz(Wbil_swz, kt, ni, QOUT / 16, lane);
    c = wmma_bf16(a, b, c);
  }
#pragma unroll
  for (int r = 0; r < 8; ++r) {
    const int mrow = r + ((lane >> 4) << 3);
    const int ncol = ni * 16 + (lane & 15);
    atomicAdd(&s_acc[mrow][ncol], c[r]);   // ds_add_f32 cross-wave reduction
  }
  __syncthreads();

  for (int idx = tid; idx < ET * QOUT; idx += 256) {
    const int e = idx >> 5, d = idx & 31;
    const int ge = e0 + e;
    if (ge < nE) x_bil[(size_t)ge * QOUT + d] = s_acc[e][d] * ssbf;
  }
}

// ---------------------------------------------------------------------------
// Up-projection: out = (silu(x@W_up_ca) + silu(x[id_swap]@W_up_ac)) * INV_SQRT2
// K=32 => one WMMA per 16x16 output tile per matrix.
// ---------------------------------------------------------------------------
#define UT 64
#define ULD (QOUT + 16)   // *2 = 96B, multiple of 16B
__global__ __launch_bounds__(256) void k_up(
    const float*  __restrict__ x_bil,          // nE x 32
    const __bf16* __restrict__ Wca_swz,        // 32 x 512, fragment-major
    const __bf16* __restrict__ Wac_swz,        // 32 x 512, fragment-major
    const int*    __restrict__ id_swap,
    float*        __restrict__ out,            // nE x 512
    int nE)
{
  __shared__ __bf16 sA[UT][ULD];               // direct rows
  __shared__ __bf16 sB[UT][ULD];               // id_swap-gathered rows

  const int tid  = threadIdx.x;
  const int wave = tid >> 5;
  const int lane = tid & 31;
  const int e0   = blockIdx.x * UT;

  for (int i = tid; i < UT * QOUT; i += 256) {
    const int r = i >> 5, cidx = i & 31;
    const int ge = e0 + r;
    float va = 0.0f, vb = 0.0f;
    if (ge < nE) {
      va = x_bil[(size_t)ge * QOUT + cidx];
      const int gs = id_swap[ge];
      vb = x_bil[(size_t)gs * QOUT + cidx];
    }
    sA[r][cidx] = (__bf16)va;
    sB[r][cidx] = (__bf16)vb;
  }
  __syncthreads();

  const int nb = wave * 64;                    // wave owns 64 output columns
  const float inv_sqrt2 = 0.70710678118654752f;
  v8f zero = {0.f, 0.f, 0.f, 0.f, 0.f, 0.f, 0.f, 0.f};

#pragma unroll
  for (int mi = 0; mi < 4; ++mi) {
    v16bf a1 = load_A_frag(&sA[mi * 16][0], ULD, lane);
    v16bf a2 = load_A_frag(&sB[mi * 16][0], ULD, lane);
#pragma unroll
    for (int ni = 0; ni < 4; ++ni) {
      const int nt = (nb >> 4) + ni;
      v16bf b1 = load_B_swz(Wca_swz, 0, nt, E_DIM / 16, lane);
      v16bf b2 = load_B_swz(Wac_swz, 0, nt, E_DIM / 16, lane);
      v8f c1 = wmma_bf16(a1, b1, zero);
      v8f c2 = wmma_bf16(a2, b2, zero);
#pragma unroll
      for (int r = 0; r < 8; ++r) {
        const int mrow = mi * 16 + r + ((lane >> 4) << 3);
        const int ncol = nb + ni * 16 + (lane & 15);
        const int ge = e0 + mrow;
        if (ge < nE)
          out[(size_t)ge * E_DIM + ncol] =
              (silu_f(c1[r]) + silu_f(c2[r])) * inv_sqrt2;
      }
    }
  }
}

// ---------------------------------------------------------------------------
// Launch
// ---------------------------------------------------------------------------
static inline size_t align256(size_t x) { return (x + 255) & ~(size_t)255; }

extern "C" void kernel_launch(void* const* d_in, const int* in_sizes, int n_in,
                              void* d_out, int out_size, void* d_ws, size_t ws_size,
                              hipStream_t stream) {
  const float* m        = (const float*)d_in[0];
  const float* rad      = (const float*)d_in[1];
  const float* cir      = (const float*)d_in[2];
  const float* srw      = (const float*)d_in[3];   // sph_rad_W1
  const float* sph      = (const float*)d_in[4];
  const float* W_db     = (const float*)d_in[5];
  const float* W_rbf    = (const float*)d_in[6];
  const float* W_down   = (const float*)d_in[7];
  const float* W_cbf    = (const float*)d_in[8];
  const float* W_bil    = (const float*)d_in[9];
  const float* W_up_ca  = (const float*)d_in[10];
  const float* W_up_ac  = (const float*)d_in[11];
  const float* s_rbf    = (const float*)d_in[12];
  const float* s_cbf    = (const float*)d_in[13];
  const float* s_sbf    = (const float*)d_in[14];
  const int*   idx_tin  = (const int*)d_in[15];
  const int*   idx_t2q  = (const int*)d_in[16];
  const int*   id_swap  = (const int*)d_in[19];
  float* out = (float*)d_out;

  const int nE = in_sizes[0] / E_DIM;

  // Workspace layout (~26.4 MB total)
  char* ws = (char*)d_ws;
  size_t off = 0;
  __bf16* wdb_swz   = (__bf16*)(ws + off); off = align256(off + (size_t)E_DIM * E_DIM * 2);
  __bf16* wrbf_swz  = (__bf16*)(ws + off); off = align256(off + (size_t)32 * E_DIM * 2);
  __bf16* wdown_swz = (__bf16*)(ws + off); off = align256(off + (size_t)E_DIM * QIN * 2);
  __bf16* wbil_swz  = (__bf16*)(ws + off); off = align256(off + (size_t)SBF * QIN * QOUT * 2);
  __bf16* wca_swz   = (__bf16*)(ws + off); off = align256(off + (size_t)QOUT * E_DIM * 2);
  __bf16* wac_swz   = (__bf16*)(ws + off); off = align256(off + (size_t)QOUT * E_DIM * 2);
  float*  x_down    = (float*)(ws + off);  off = align256(off + (size_t)nE * QIN * 4);
  float*  x_bil     = (float*)(ws + off);  off = align256(off + (size_t)nE * QOUT * 4);
  (void)ws_size;

  // 1) bf16 weight conversion + fragment-major swizzle (Wrbf K-padded 16->32)
  {
    int n;
    n = E_DIM * E_DIM;    k_swz<<<(n + 255) / 256, 256, 0, stream>>>(W_db,    wdb_swz,   E_DIM,     E_DIM, E_DIM);
    n = 32 * E_DIM;       k_swz<<<(n + 255) / 256, 256, 0, stream>>>(W_rbf,   wrbf_swz,  RBF,       E_DIM, 32);
    n = E_DIM * QIN;      k_swz<<<(n + 255) / 256, 256, 0, stream>>>(W_down,  wdown_swz, E_DIM,     QIN,   E_DIM);
    n = SBF * QIN * QOUT; k_swz<<<(n + 255) / 256, 256, 0, stream>>>(W_bil,   wbil_swz,  SBF * QIN, QOUT,  SBF * QIN);
    n = QOUT * E_DIM;     k_swz<<<(n + 255) / 256, 256, 0, stream>>>(W_up_ca, wca_swz,   QOUT,      E_DIM, QOUT);
    n = QOUT * E_DIM;     k_swz<<<(n + 255) / 256, 256, 0, stream>>>(W_up_ac, wac_swz,   QOUT,      E_DIM, QOUT);
  }

  // 2) fused edge MLP
  {
    dim3 grid((nE + MT - 1) / MT);
    k_edge<<<grid, 256, 0, stream>>>(m, rad, wdb_swz, wrbf_swz, wdown_swz,
                                     s_rbf, x_down, nE);
  }

  // 3) quad interaction + bilinear
  {
    dim3 grid((nE + ET - 1) / ET);
    k_quad<<<grid, 256, 0, stream>>>(x_down, cir, W_cbf, sph, srw, wbil_swz,
                                     s_cbf, s_sbf, idx_tin, idx_t2q, x_bil, nE);
  }

  // 4) up-projection + swap + combine
  {
    dim3 grid((nE + UT - 1) / UT);
    k_up<<<grid, 256, 0, stream>>>(x_bil, wca_swz, wac_swz, id_swap, out, nE);
  }

  (void)n_in; (void)out_size;
}
